// Model_26834955665875
// MI455X (gfx1250) — matrix operand order repaired and checked by reading, not compiled, not run on hardware
//
#include <hip/hip_runtime.h>
#include <hip/hip_bf16.h>
#include <cmath>

typedef _Float16 f16;
typedef __attribute__((ext_vector_type(16))) _Float16 v16h;
typedef __attribute__((ext_vector_type(8)))  _Float16 v8h;
typedef __attribute__((ext_vector_type(8)))  float    v8f;

// ---------------------------------------------------------------------------
// WMMA f16 GEMM:  C[M,N] = A[M,K] * Bt[N,K]^T   (K multiple of 32, zero-padded)
// One wave (32 threads) computes one 16x16 tile. f32 accumulate.
// A fragment = contiguous 16 halfs per lane half (lane -> M row), B fragment = contiguous 16
// halfs per lane half (lane -> N column), loaded from row-major Bt.
// Out-of-range A rows / B cols are CLAMPED (not zeroed): a clamped A row only
// pollutes D rows >= M and a clamped B col only pollutes D cols >= N, neither
// of which is ever stored -> branch-free, clause-friendly inner loop.
// ---------------------------------------------------------------------------
__global__ void wmma_gemm_f16(const f16* __restrict__ A, int lda, long long strideA,
                              const f16* __restrict__ Bt, int ldb, long long strideB,
                              float* __restrict__ C, int ldc, long long strideC,
                              int M, int N, int K,
                              const float* __restrict__ biasRow,
                              const float* __restrict__ biasCol0,
                              const float* __restrict__ biasCol1)
{
    const int bz = blockIdx.z;
    A  += (long long)bz * strideA;
    Bt += (long long)bz * strideB;
    C  += (long long)bz * strideC;

    const int tm   = blockIdx.y << 4;
    const int tn   = blockIdx.x << 4;
    const int lane = threadIdx.x & 31;
    const int hi   = lane >> 4;      // which half of the wave
    const int r    = lane & 15;

    int arow = tm + r; if (arow >= M) arow = M - 1;   // clamp (see note above)
    int brow = tn + r; if (brow >= N) brow = N - 1;

    const f16* ap = A  + (long long)arow * lda + 16 * hi;
    const f16* bp = Bt + (long long)brow * ldb + 16 * hi;

    v8f acc = {};
    for (int k0 = 0; k0 < K; k0 += 32) {
        v8h a0 = *(const v8h*)(ap + k0);        // K = k0 + 16*hi + [0..7]
        v8h a1 = *(const v8h*)(ap + k0 + 8);    // K = k0 + 16*hi + 8 + [0..7]
        v16h b = *(const v16h*)(bp + k0);       // K = k0 + 16*hi + [0..15]
        v16h a;
#pragma unroll
        for (int e = 0; e < 8; ++e) { a[e] = a0[e]; a[e + 8] = a1[e]; }
        acc = __builtin_amdgcn_wmma_f32_16x16x32_f16(false, a, false, b,
                                                     (short)0, acc, false, false);
    }

    const int col = tn + r;
    if (col < N) {
        float cb = 0.0f;
        if (biasCol0) cb += biasCol0[col];
        if (biasCol1) cb += biasCol1[col];
#pragma unroll
        for (int j = 0; j < 8; ++j) {
            int row = tm + 8 * hi + j;              // D layout: VGPR j -> M = 8*hi + j
            if (row < M) {
                float v = acc[j] + cb;
                if (biasRow) v += biasRow[row];
                C[(long long)row * ldc + col] = v;
            }
        }
    }
}

// ---------------------------------------------------------------------------
// small helpers
// ---------------------------------------------------------------------------
__global__ void add_kernel(const float* __restrict__ a, const float* __restrict__ b,
                           float* __restrict__ c, int n)
{
    int i = blockIdx.x * 256 + threadIdx.x;
    if (i < n) c[i] = a[i] + b[i];
}

__global__ void cvt_f16_pad(const float* __restrict__ src, f16* __restrict__ dst,
                            int rows, int cols, int ld)
{
    long long i   = (long long)blockIdx.x * 256 + threadIdx.x;
    long long tot = (long long)rows * ld;
    if (i >= tot) return;
    int rr = (int)(i / ld), cc = (int)(i % ld);
    dst[i] = (cc < cols) ? (f16)src[(long long)rr * cols + cc] : (f16)0.0f;
}

// ---------------------------------------------------------------------------
// data BN (per m*V*C channel over N,T) + layout to gcn1 input (NM,C,T,V)
// ---------------------------------------------------------------------------
__global__ void databn_kernel(const float* __restrict__ x, const float* __restrict__ g,
                              const float* __restrict__ b, float* __restrict__ hin1)
{
    int ch = blockIdx.x;                 // ch = m*V*C + v*C + c  (150)
    int m = ch / 75, v = (ch / 3) % 25, c = ch % 3;
    __shared__ float ss[64], sq[64];
    float s0 = 0.f, s1 = 0.f;
    for (int i = threadIdx.x; i < 200; i += 64) {
        int n = i / 100, t = i % 100;
        float val = x[(((n * 3 + c) * 100 + t) * 25 + v) * 2 + m];
        s0 += val; s1 += val * val;
    }
    ss[threadIdx.x] = s0; sq[threadIdx.x] = s1; __syncthreads();
    for (int off = 32; off > 0; off >>= 1) {
        if (threadIdx.x < off) { ss[threadIdx.x] += ss[threadIdx.x + off];
                                 sq[threadIdx.x] += sq[threadIdx.x + off]; }
        __syncthreads();
    }
    float mean = ss[0] * (1.0f / 200.0f);
    float var  = sq[0] * (1.0f / 200.0f) - mean * mean;
    float rs   = rsqrtf(var + 1e-5f);
    float gg = g[ch], bb = b[ch];
    for (int i = threadIdx.x; i < 200; i += 64) {
        int n = i / 100, t = i % 100;
        float val = x[(((n * 3 + c) * 100 + t) * 25 + v) * 2 + m];
        int nm = n * 2 + m;
        hin1[((nm * 3 + c) * 100 + t) * 25 + v] = (val - mean) * rs * gg + bb;
    }
}

// ---------------------------------------------------------------------------
// support (adjacency contraction), written transposed+f16 as GEMM Bt operand
// st1[n][p=(t,v)][kc] ,  kc = k*3+c  (padded to 64)
// ---------------------------------------------------------------------------
__global__ void build_st1(const float* __restrict__ A1, const float* __restrict__ hin1,
                          f16* __restrict__ st)
{
    long long i = (long long)blockIdx.x * 256 + threadIdx.x;
    if (i >= 4LL * 2500 * 64) return;
    int kc = (int)(i % 64);
    long long np = i / 64;
    int p = (int)(np % 2500), n = (int)(np / 2500);
    if (kc >= 39) { st[i] = (f16)0.0f; return; }
    int k = kc / 3, c = kc % 3;
    int t = p / 25, v = p % 25;
    const float* ar = A1 + (k * 25 + v) * 25;
    const float* hr = hin1 + ((n * 3 + c) * 100 + t) * 25;
    float acc = 0.f;
#pragma unroll
    for (int u = 0; u < 25; ++u) acc += ar[u] * hr[u];
    st[i] = (f16)acc;
}

// st2[n][p=(s,v)][kc] , kc = k*96+c , K = 1248 (already mult of 32)
__global__ void build_st2(const float* __restrict__ A2, const float* __restrict__ hin2,
                          f16* __restrict__ st)
{
    long long i = (long long)blockIdx.x * 256 + threadIdx.x;
    if (i >= 4LL * 1250 * 1248) return;
    int kc = (int)(i % 1248);
    long long np = i / 1248;
    int p = (int)(np % 1250), n = (int)(np / 1250);
    int k = kc / 96, c = kc % 96;
    int s = p / 25, v = p % 25;
    const float* ar = A2 + (k * 25 + v) * 25;
    const float* hr = hin2 + ((n * 96 + c) * 50 + s) * 25;
    float acc = 0.f;
#pragma unroll
    for (int u = 0; u < 25; ++u) acc += ar[u] * hr[u];
    st[i] = (f16)acc;
}

// ---------------------------------------------------------------------------
// per-channel BN (+ReLU) kernels, fused with the inter-stage transposes
// ---------------------------------------------------------------------------
__global__ void bn_gcn1(const float* __restrict__ src, const float* __restrict__ g,
                        const float* __restrict__ bta, float* __restrict__ seq1)
{
    int c = blockIdx.x;                          // 0..95
    __shared__ float ss[256], sq[256];
    float s0 = 0.f, s1 = 0.f;
    for (int i = threadIdx.x; i < 10000; i += 256) {
        int n = i / 2500, p = i % 2500;
        float v = src[(n * 96 + c) * 2500 + p];
        s0 += v; s1 += v * v;
    }
    ss[threadIdx.x] = s0; sq[threadIdx.x] = s1; __syncthreads();
    for (int off = 128; off > 0; off >>= 1) {
        if (threadIdx.x < off) { ss[threadIdx.x] += ss[threadIdx.x + off];
                                 sq[threadIdx.x] += sq[threadIdx.x + off]; }
        __syncthreads();
    }
    float mean = ss[0] * (1.0f / 10000.0f);
    float var  = sq[0] * (1.0f / 10000.0f) - mean * mean;
    float rs = rsqrtf(var + 1e-5f);
    float gg = g[c], bb = bta[c];
    for (int i = threadIdx.x; i < 10000; i += 256) {
        int n = i / 2500, p = i % 2500;
        float v = src[(n * 96 + c) * 2500 + p];
        v = (v - mean) * rs * gg + bb;
        v = v > 0.f ? v : 0.f;
        int t = p / 25, vv = p % 25;
        seq1[((n * 25 + vv) * 100 + t) * 96 + c] = v;  // (NM*V, T, c1)
    }
}

__global__ void bn_seg1(const float* __restrict__ src, const float* __restrict__ g,
                        const float* __restrict__ bta, float* __restrict__ hin2)
{
    int s = blockIdx.x;                          // 0..49
    __shared__ float ss[256], sq[256];
    float s0 = 0.f, s1 = 0.f;
    for (int i = threadIdx.x; i < 9600; i += 256) {
        int b = i / 96, k = i % 96;
        float v = src[(b * 50 + s) * 96 + k];
        s0 += v; s1 += v * v;
    }
    ss[threadIdx.x] = s0; sq[threadIdx.x] = s1; __syncthreads();
    for (int off = 128; off > 0; off >>= 1) {
        if (threadIdx.x < off) { ss[threadIdx.x] += ss[threadIdx.x + off];
                                 sq[threadIdx.x] += sq[threadIdx.x + off]; }
        __syncthreads();
    }
    float mean = ss[0] * (1.0f / 9600.0f);
    float var  = sq[0] * (1.0f / 9600.0f) - mean * mean;
    float rs = rsqrtf(var + 1e-5f);
    float gg = g[s], bb = bta[s];
    for (int i = threadIdx.x; i < 9600; i += 256) {
        int b = i / 96, k = i % 96;
        float v = src[(b * 50 + s) * 96 + k];
        v = (v - mean) * rs * gg + bb;
        hin2[(((b / 25) * 96 + k) * 50 + s) * 25 + (b % 25)] = v;  // (NM,c1,S1,V)
    }
}

__global__ void bn_gcn2(const float* __restrict__ src, const float* __restrict__ g,
                        const float* __restrict__ bta, float* __restrict__ seq2)
{
    int c = blockIdx.x;                          // 0..191
    __shared__ float ss[256], sq[256];
    float s0 = 0.f, s1 = 0.f;
    for (int i = threadIdx.x; i < 5000; i += 256) {
        int n = i / 1250, p = i % 1250;
        float v = src[(n * 192 + c) * 1250 + p];
        s0 += v; s1 += v * v;
    }
    ss[threadIdx.x] = s0; sq[threadIdx.x] = s1; __syncthreads();
    for (int off = 128; off > 0; off >>= 1) {
        if (threadIdx.x < off) { ss[threadIdx.x] += ss[threadIdx.x + off];
                                 sq[threadIdx.x] += sq[threadIdx.x + off]; }
        __syncthreads();
    }
    float mean = ss[0] * (1.0f / 5000.0f);
    float var  = sq[0] * (1.0f / 5000.0f) - mean * mean;
    float rs = rsqrtf(var + 1e-5f);
    float gg = g[c], bb = bta[c];
    for (int i = threadIdx.x; i < 5000; i += 256) {
        int n = i / 1250, p = i % 1250;
        float v = src[(n * 192 + c) * 1250 + p];
        v = (v - mean) * rs * gg + bb;
        v = v > 0.f ? v : 0.f;
        int s = p / 25, vv = p % 25;
        seq2[((n * 25 + vv) * 50 + s) * 192 + c] = v;  // (NM*V, S1, c2)
    }
}

__global__ void bn_seg2(const float* __restrict__ src, const float* __restrict__ g,
                        const float* __restrict__ bta, float* __restrict__ dst)
{
    int s = blockIdx.x;                          // 0..29
    __shared__ float ss[256], sq[256];
    float s0 = 0.f, s1 = 0.f;
    for (int i = threadIdx.x; i < 19200; i += 256) {
        int b = i / 192, k = i % 192;
        float v = src[(b * 30 + s) * 192 + k];
        s0 += v; s1 += v * v;
    }
    ss[threadIdx.x] = s0; sq[threadIdx.x] = s1; __syncthreads();
    for (int off = 128; off > 0; off >>= 1) {
        if (threadIdx.x < off) { ss[threadIdx.x] += ss[threadIdx.x + off];
                                 sq[threadIdx.x] += sq[threadIdx.x + off]; }
        __syncthreads();
    }
    float mean = ss[0] * (1.0f / 19200.0f);
    float var  = sq[0] * (1.0f / 19200.0f) - mean * mean;
    float rs = rsqrtf(var + 1e-5f);
    float gg = g[s], bb = bta[s];
    for (int i = threadIdx.x; i < 19200; i += 256) {
        int b = i / 192, k = i % 192;
        float v = src[(b * 30 + s) * 192 + k];
        dst[(b * 30 + s) * 192 + k] = (v - mean) * rs * gg + bb;
    }
}

// ---------------------------------------------------------------------------
// log-signature features, one block per (b,s) row; writes f16 GEMM rows
// (numpy linspace + banker's rounding reproduced with nearbyint)
// ---------------------------------------------------------------------------
__device__ __forceinline__ int tvec_dev(int k, int T, int S)
{
    if (k == S) return T;
    double delta = (double)(T - 1) / (double)S;      // matches np.linspace step
    return (int)nearbyint(1.0 + delta * (double)k);  // round-half-even
}

__global__ void logsig_feats_kernel(const float* __restrict__ seq, f16* __restrict__ feats,
                                    int r0, int T, int d, int S, int L, int ldF)
{
    __shared__ float pts[3 * 192];
    __shared__ float rel[2 * 192];
    __shared__ float dxs[2 * 192];
    int gr = r0 + blockIdx.x;
    int b = gr / S, s = gr % S;
    int st = tvec_dev(s, T, S) - 1;
    int en = tvec_dev(s + 1, T, S) - 1;
    const float* sb = seq + (long long)b * T * d;
    for (int i = threadIdx.x; i < L * d; i += blockDim.x) {
        int l = i / d, c = i - l * d;
        int t = st + l; if (t > en) t = en;          // idx = min(start+l, end)
        pts[i] = sb[(long long)t * d + c];
    }
    __syncthreads();
    for (int i = threadIdx.x; i < (L - 1) * d; i += blockDim.x) {
        int l = i / d, c = i - l * d;
        rel[i] = pts[l * d + c] - pts[c];
        dxs[i] = pts[(l + 1) * d + c] - pts[l * d + c];
    }
    __syncthreads();
    f16* fr = feats + (long long)blockIdx.x * ldF;
    for (int c = threadIdx.x; c < d; c += blockDim.x) {
        fr[c]     = (f16)pts[c];                                // x at segment start
        fr[d + c] = (f16)(pts[(L - 1) * d + c] - pts[c]);       // increment
    }
    int base = 2 * d, npairs = d * (d - 1) / 2;
    for (int i = threadIdx.x; i < d - 1; i += blockDim.x) {
        long long off = base + (long long)i * (d - 1) - (long long)i * (i - 1) / 2 - i - 1;
        for (int j = i + 1; j < d; ++j) {
            float acc = 0.0f;
            for (int l = 0; l < L - 1; ++l)
                acc += rel[l * d + i] * dxs[l * d + j] - rel[l * d + j] * dxs[l * d + i];
            fr[off + j] = (f16)(0.5f * acc);                    // Levy area
        }
    }
    for (int c = base + npairs + threadIdx.x; c < ldF; c += blockDim.x)
        fr[c] = (f16)0.0f;                                       // K zero-pad
}

// ---------------------------------------------------------------------------
// LSTM scan: one block per independent batch row; h/c/gates in LDS
// ---------------------------------------------------------------------------
__global__ void lstm_kernel(const float* __restrict__ xp, const float* __restrict__ Whh,
                            float* __restrict__ ys, int S, int H)
{
    __shared__ float hs[192], cs[192], gates[768];
    int b = blockIdx.x;
    for (int k = threadIdx.x; k < H; k += blockDim.x) { hs[k] = 0.f; cs[k] = 0.f; }
    __syncthreads();
    for (int t = 0; t < S; ++t) {
        const float* xr = xp + ((long long)b * S + t) * 4 * H;
        for (int j = threadIdx.x; j < 4 * H; j += blockDim.x) {
            const float* w = Whh + (long long)j * H;
            float acc = xr[j];
            for (int k = 0; k < H; ++k) acc += w[k] * hs[k];
            gates[j] = acc;
        }
        __syncthreads();
        for (int k = threadIdx.x; k < H; k += blockDim.x) {
            float ig = 1.0f / (1.0f + expf(-gates[k]));
            float fg = 1.0f / (1.0f + expf(-gates[H + k]));
            float gg = tanhf(gates[2 * H + k]);
            float og = 1.0f / (1.0f + expf(-gates[3 * H + k]));
            float c = fg * cs[k] + ig * gg;
            float h = og * tanhf(c);
            cs[k] = c; hs[k] = h;
            ys[((long long)b * S + t) * H + k] = h;
        }
        __syncthreads();
    }
}

// ---------------------------------------------------------------------------
// pooling + FC
// ---------------------------------------------------------------------------
__global__ void pool_kernel(const float* __restrict__ ys2bn, float* __restrict__ pooled)
{
    int n = blockIdx.x / 192, ch = blockIdx.x % 192;
    __shared__ float ss[128];
    float s = 0.f;
    for (int i = threadIdx.x; i < 1500; i += 128) {        // m(2)*v(25)*s(30)
        int m = i / 750, rem = i % 750, v = rem / 30, s2 = rem % 30;
        s += ys2bn[(((n * 2 + m) * 25 + v) * 30 + s2) * 192 + ch];
    }
    ss[threadIdx.x] = s; __syncthreads();
    for (int off = 64; off > 0; off >>= 1) {
        if (threadIdx.x < off) ss[threadIdx.x] += ss[threadIdx.x + off];
        __syncthreads();
    }
    if (threadIdx.x == 0) pooled[n * 192 + ch] = ss[0] * (1.0f / 1500.0f);
}

__global__ void fc_kernel(const float* __restrict__ pooled, const float* __restrict__ fcW,
                          const float* __restrict__ fcb, float* __restrict__ out)
{
    int tid = blockIdx.x * blockDim.x + threadIdx.x;
    if (tid >= 120) return;
    int n = tid / 60, o = tid % 60;
    float acc = fcb[o];
    for (int ch = 0; ch < 192; ++ch) acc += pooled[n * 192 + ch] * fcW[o * 192 + ch];
    out[n * 60 + o] = acc;
}

// ---------------------------------------------------------------------------
// host side
// ---------------------------------------------------------------------------
static inline int tvec_host(int k, int T, int S)
{
    if (k == S) return T;
    double delta = (double)(T - 1) / (double)S;
    return (int)std::nearbyint(1.0 + delta * (double)k);
}

extern "C" void kernel_launch(void* const* d_in, const int* in_sizes, int n_in,
                              void* d_out, int out_size, void* d_ws, size_t ws_size,
                              hipStream_t stream)
{
    (void)in_sizes; (void)n_in; (void)out_size; (void)ws_size;
    const float* x         = (const float*)d_in[0];
    const float* data_bn_g = (const float*)d_in[2];
    const float* data_bn_b = (const float*)d_in[3];
    const float* Ap1       = (const float*)d_in[4];
    const float* Ar1       = (const float*)d_in[5];
    const float* W1        = (const float*)d_in[6];
    const float* b1        = (const float*)d_in[7];
    const float* bn1_g     = (const float*)d_in[8];
    const float* bn1_b     = (const float*)d_in[9];
    const float* Wih1      = (const float*)d_in[10];
    const float* Whh1      = (const float*)d_in[11];
    const float* bih1      = (const float*)d_in[12];
    const float* bhh1      = (const float*)d_in[13];
    const float* bnseg1_g  = (const float*)d_in[14];
    const float* bnseg1_b  = (const float*)d_in[15];
    const float* Ap2       = (const float*)d_in[16];
    const float* Ar2       = (const float*)d_in[17];
    const float* W2        = (const float*)d_in[18];
    const float* b2        = (const float*)d_in[19];
    const float* bn2_g     = (const float*)d_in[20];
    const float* bn2_b     = (const float*)d_in[21];
    const float* Wih2      = (const float*)d_in[22];
    const float* Whh2      = (const float*)d_in[23];
    const float* bih2      = (const float*)d_in[24];
    const float* bhh2      = (const float*)d_in[25];
    const float* bnseg2_g  = (const float*)d_in[26];
    const float* bnseg2_b  = (const float*)d_in[27];
    const float* fcW       = (const float*)d_in[28];
    const float* fcb       = (const float*)d_in[29];
    float* out = (float*)d_out;

    // bump allocator over d_ws (all buffers written before read)
    char* w = (char*)d_ws;
    auto alloc = [&](size_t bytes) -> void* {
        void* p = (void*)w;
        w += (bytes + 255) & ~(size_t)255;
        return p;
    };
    float* A1       = (float*)alloc(8125 * 4);
    float* A2       = (float*)alloc(8125 * 4);
    float* hin1     = (float*)alloc(30000 * 4);
    float* gcnpre1  = (float*)alloc(960000 * 4);      // (4,96,2500)
    float* seq1     = (float*)alloc(960000 * 4);      // (100,100,96)
    float* xp1      = (float*)alloc(1920000 * 4);     // (100,50,384)
    float* ys1      = (float*)alloc(480000 * 4);      // (100,50,96)
    float* hin2     = (float*)alloc(480000 * 4);      // (4,96,50,25)
    float* gcnpre2  = (float*)alloc(960000 * 4);      // (4,192,1250)
    float* seq2     = (float*)alloc(960000 * 4);      // (100,50,192)
    float* xp2      = (float*)alloc(2304000 * 4);     // (100,30,768)
    float* ys2      = (float*)alloc(576000 * 4);      // (100,30,192)
    float* ys2bn    = (float*)alloc(576000 * 4);
    float* pooled   = (float*)alloc(384 * 4);
    f16* W1f16      = (f16*)alloc(96LL * 64 * 2);
    f16* st1        = (f16*)alloc(4LL * 2500 * 64 * 2);
    f16* Wih1f16    = (f16*)alloc(384LL * 4768 * 2);
    f16* W2f16      = (f16*)alloc(192LL * 1248 * 2);
    f16* st2        = (f16*)alloc(4LL * 1250 * 1248 * 2);
    f16* Wih2f16    = (f16*)alloc(768LL * 18720 * 2);
    f16* featsChunk = (f16*)alloc(512LL * 18720 * 2); // shared for both layers

    // segment length L (same computation as the reference, static dims only)
    int L1 = 0;
    for (int s = 0; s < 50; ++s) {
        int st = tvec_host(s, 100, 50), en = tvec_host(s + 1, 100, 50);
        if (en - st + 1 > L1) L1 = en - st + 1;
    }
    int L2 = 0;
    for (int s = 0; s < 30; ++s) {
        int st = tvec_host(s, 50, 30), en = tvec_host(s + 1, 50, 30);
        if (en - st + 1 > L2) L2 = en - st + 1;
    }

    // ---- stage 0: adjacency sums, data BN, weight staging ----
    add_kernel<<<32, 256, 0, stream>>>(Ap1, Ar1, A1, 8125);
    add_kernel<<<32, 256, 0, stream>>>(Ap2, Ar2, A2, 8125);
    databn_kernel<<<150, 64, 0, stream>>>(x, data_bn_g, data_bn_b, hin1);
    cvt_f16_pad<<<(96 * 64 + 255) / 256, 256, 0, stream>>>(W1, W1f16, 96, 39, 64);
    cvt_f16_pad<<<(int)((384LL * 4768 + 255) / 256), 256, 0, stream>>>(Wih1, Wih1f16, 384, 4752, 4768);
    cvt_f16_pad<<<(int)((192LL * 1248 + 255) / 256), 256, 0, stream>>>(W2, W2f16, 192, 1248, 1248);
    cvt_f16_pad<<<(int)((768LL * 18720 + 255) / 256), 256, 0, stream>>>(Wih2, Wih2f16, 768, 18720, 18720);

    // ---- MS-GCN layer 1 ----
    build_st1<<<(int)((4LL * 2500 * 64 + 255) / 256), 256, 0, stream>>>(A1, hin1, st1);
    wmma_gemm_f16<<<dim3(157, 6, 4), 32, 0, stream>>>(
        W1f16, 64, 0LL, st1, 64, 2500LL * 64, gcnpre1, 2500, 96LL * 2500,
        96, 2500, 64, b1, nullptr, nullptr);
    bn_gcn1<<<96, 256, 0, stream>>>(gcnpre1, bn1_g, bn1_b, seq1);

    // ---- logsig features + LSTM input projection, layer 1 (chunked) ----
    const int CH = 512;
    for (int r0 = 0; r0 < 5000; r0 += CH) {
        int rows = (5000 - r0 < CH) ? (5000 - r0) : CH;
        logsig_feats_kernel<<<rows, 128, 0, stream>>>(seq1, featsChunk, r0, 100, 96, 50, L1, 4768);
        wmma_gemm_f16<<<dim3(24, (rows + 15) / 16, 1), 32, 0, stream>>>(
            featsChunk, 4768, 0LL, Wih1f16, 4768, 0LL,
            xp1 + (long long)r0 * 384, 384, 0LL,
            rows, 384, 4768, nullptr, bih1, bhh1);
    }
    lstm_kernel<<<100, 256, 0, stream>>>(xp1, Whh1, ys1, 50, 96);
    bn_seg1<<<50, 256, 0, stream>>>(ys1, bnseg1_g, bnseg1_b, hin2);

    // ---- MS-GCN layer 2 ----
    build_st2<<<(int)((4LL * 1250 * 1248 + 255) / 256), 256, 0, stream>>>(A2, hin2, st2);
    wmma_gemm_f16<<<dim3(79, 12, 4), 32, 0, stream>>>(
        W2f16, 1248, 0LL, st2, 1248, 1250LL * 1248, gcnpre2, 1250, 192LL * 1250,
        192, 1250, 1248, b2, nullptr, nullptr);
    bn_gcn2<<<192, 256, 0, stream>>>(gcnpre2, bn2_g, bn2_b, seq2);

    // ---- logsig features + LSTM input projection, layer 2 (chunked) ----
    for (int r0 = 0; r0 < 3000; r0 += CH) {
        int rows = (3000 - r0 < CH) ? (3000 - r0) : CH;
        logsig_feats_kernel<<<rows, 128, 0, stream>>>(seq2, featsChunk, r0, 50, 192, 30, L2, 18720);
        wmma_gemm_f16<<<dim3(48, (rows + 15) / 16, 1), 32, 0, stream>>>(
            featsChunk, 18720, 0LL, Wih2f16, 18720, 0LL,
            xp2 + (long long)r0 * 768, 768, 0LL,
            rows, 768, 18720, nullptr, bih2, bhh2);
    }
    lstm_kernel<<<100, 256, 0, stream>>>(xp2, Whh2, ys2, 30, 192);
    bn_seg2<<<30, 256, 0, stream>>>(ys2, bnseg2_g, bnseg2_b, ys2bn);

    // ---- pool + FC ----
    pool_kernel<<<384, 128, 0, stream>>>(ys2bn, pooled);
    fc_kernel<<<1, 128, 0, stream>>>(pooled, fcW, fcb, out);
}
